// SelfAttention_51685636440170
// MI455X (gfx1250) — matrix operand order repaired and checked by reading, not compiled, not run on hardware
//
#include <hip/hip_runtime.h>
#include <hip/hip_bf16.h>

#define DEVFN __device__ __forceinline__
#define AS1 __attribute__((address_space(1)))
#define AS3 __attribute__((address_space(3)))

namespace {
constexpr int kB = 8, kS = 2048, kD = 1024, kDK = 32;
constexpr float kNinf = -1e24f;
constexpr float kInvSqrtDk = 0.17677669529663689f; // 1/sqrt(32)
}

typedef __attribute__((ext_vector_type(16))) __bf16 bf16x16;
typedef __attribute__((ext_vector_type(8)))  __bf16 bf16x8;
typedef __attribute__((ext_vector_type(8)))  float  f32x8;
typedef __attribute__((ext_vector_type(4)))  int    int4v;

// Native converts -> backend can use v_cvt_pk_bf16_f32 (RTNE, same as ref).
DEVFN __bf16 f2bf(float f) { return (__bf16)f; }
DEVFN unsigned short f2bfu(float f) {
  return __builtin_bit_cast(unsigned short, (__bf16)f);
}
DEVFN __bf16 bfbits(unsigned short s) { return __builtin_bit_cast(__bf16, s); }

DEVFN f32x8 zero8() {
  f32x8 r;
#pragma unroll
  for (int j = 0; j < 8; ++j) r[j] = 0.0f;
  return r;
}
DEVFN bf16x16 zbf16() {
  bf16x16 r;
#pragma unroll
  for (int j = 0; j < 16; ++j) r[j] = bfbits(0);
  return r;
}

DEVFN bf16x8 cvt8(const float* p) {
  float4 a = *(const float4*)p;
  float4 b = *(const float4*)(p + 4);
  bf16x8 r;
  r[0] = f2bf(a.x); r[1] = f2bf(a.y); r[2] = f2bf(a.z); r[3] = f2bf(a.w);
  r[4] = f2bf(b.x); r[5] = f2bf(b.y); r[6] = f2bf(b.z); r[7] = f2bf(b.w);
  return r;
}
DEVFN bf16x16 cat16(bf16x8 lo, bf16x8 hi) {
  return __builtin_shufflevector(lo, hi, 0,1,2,3,4,5,6,7,8,9,10,11,12,13,14,15);
}
DEVFN bf16x8  ld8 (const unsigned short* p) { return *(const bf16x8*)p; }
DEVFN bf16x16 ld16(const unsigned short* p) { return *(const bf16x16*)p; }

DEVFN f32x8 wmma_bf16(bf16x16 a, bf16x16 b, f32x8 c) {
  return __builtin_amdgcn_wmma_f32_16x16x32_bf16(false, a, false, b, (short)0, c,
                                                 false, false);
}

// ---- async global->LDS (16B per lane), with synchronous fallback ----------
DEVFN void async_ld16(const void* g, void* l) {
#if __has_builtin(__builtin_amdgcn_global_load_async_to_lds_b128)
  __builtin_amdgcn_global_load_async_to_lds_b128((AS1 int4v*)g, (AS3 int4v*)l,
                                                 0, 0);
#else
  *(uint4*)l = *(const uint4*)g;      // synchronous fallback
#endif
}
DEVFN void async_wait0() {
#if __has_builtin(__builtin_amdgcn_s_wait_asynccnt)
  __builtin_amdgcn_s_wait_asynccnt(0);
#else
  asm volatile("s_wait_asynccnt 0x0" ::: "memory");
#endif
}

// ---------------- prep: transpose weight to bf16 (dst[n][k] = src[k][n]) ----
__global__ void transpose_to_bf16(const float* __restrict__ src,
                                  unsigned short* __restrict__ dst,
                                  int K, int N) {
  int total = K * N;
  for (int i = blockIdx.x * blockDim.x + threadIdx.x; i < total;
       i += gridDim.x * blockDim.x) {
    int k = i / N, n = i - k * N;
    dst[(size_t)n * K + k] = f2bfu(src[i]);
  }
}

// ---------------- Q/K projection: [16-token strip] x [dk=32] per wave -------
__global__ void proj_qk_kernel(const float* __restrict__ x,
                               const unsigned short* __restrict__ WqT,
                               const unsigned short* __restrict__ WkT,
                               const unsigned char* __restrict__ kpm,
                               unsigned short* __restrict__ Q,
                               unsigned short* __restrict__ K) {
  const int lane = threadIdx.x & 31, wave = threadIdx.x >> 5;
  const int strip = blockIdx.x * 8 + wave;        // 0..1023 (16 tokens each)
  const int tbase = strip * 16;
  const int hp = lane >> 4, ln = lane & 15;
  const int arow  = tbase + ln;
  const int abase = hp ? 16 : 0;
  const int bkoff = hp ? 16 : 0;
  f32x8 aq0 = zero8(), aq1 = zero8(), ak0 = zero8(), ak1 = zero8();
  for (int kb = 0; kb < kD / 32; ++kb) {
    const float* xr = x + (size_t)arow * kD + kb * 32;
    bf16x16 A = cat16(cvt8(xr + abase), cvt8(xr + abase + 8));
    bf16x16 Bq0 = ld16(WqT + (size_t)(ln     ) * kD + kb * 32 + bkoff);
    bf16x16 Bq1 = ld16(WqT + (size_t)(ln + 16) * kD + kb * 32 + bkoff);
    bf16x16 Bk0 = ld16(WkT + (size_t)(ln     ) * kD + kb * 32 + bkoff);
    bf16x16 Bk1 = ld16(WkT + (size_t)(ln + 16) * kD + kb * 32 + bkoff);
    aq0 = wmma_bf16(A, Bq0, aq0);
    aq1 = wmma_bf16(A, Bq1, aq1);
    ak0 = wmma_bf16(A, Bk0, ak0);
    ak1 = wmma_bf16(A, Bk1, ak1);
  }
  const unsigned short ninf = f2bfu(kNinf);
#pragma unroll
  for (int i = 0; i < 8; ++i) {
    int t = tbase + i + hp * 8;
    bool pad = kpm[t] != 0;   // reference masks K *values*, not scores
    Q[(size_t)t * kDK + ln]      = f2bfu(aq0[i]);
    Q[(size_t)t * kDK + 16 + ln] = f2bfu(aq1[i]);
    K[(size_t)t * kDK + ln]      = pad ? ninf : f2bfu(ak0[i]);
    K[(size_t)t * kDK + 16 + ln] = pad ? ninf : f2bfu(ak1[i]);
  }
}

// ---------------- V projection, stored transposed: VT[b][d][s] -------------
__global__ void proj_v_kernel(const float* __restrict__ x,
                              const unsigned short* __restrict__ WvT,
                              unsigned short* __restrict__ VT) {
  const int lane = threadIdx.x & 31, wave = threadIdx.x >> 5;
  const int d0 = blockIdx.x * 128 + wave * 16;    // 16-row d strip per wave
  const int tbase = blockIdx.y * 128;             // 128 tokens per block
  const int hp = lane >> 4, ln = lane & 15;
  const int abase = hp ? 16 : 0;
  const int bkoff = hp ? 16 : 0;
  f32x8 acc[8];
#pragma unroll
  for (int t = 0; t < 8; ++t) acc[t] = zero8();
  for (int kb = 0; kb < kD / 32; ++kb) {
    const unsigned short* wr = WvT + (size_t)(d0 + ln) * kD + kb * 32;
    bf16x16 A = cat16(ld8(wr + abase), ld8(wr + abase + 8));
#pragma unroll
    for (int tt = 0; tt < 8; ++tt) {
      const float* xr = x + (size_t)(tbase + tt * 16 + ln) * kD + kb * 32 + bkoff;
      bf16x16 Bx = cat16(cvt8(xr), cvt8(xr + 8));
      acc[tt] = wmma_bf16(A, Bx, acc[tt]);
    }
  }
#pragma unroll
  for (int tt = 0; tt < 8; ++tt) {
#pragma unroll
    for (int i = 0; i < 8; ++i) {
      int d = d0 + i + hp * 8;
      int t = tbase + tt * 16 + ln;
      int b = t >> 11, s = t & (kS - 1);
      VT[((size_t)b * kD + d) * kS + s] = f2bfu(acc[tt][i]);
    }
  }
}

// ---------------- flash attention: 64 queries / WG, 8 waves ----------------
// waves 0-3: S = QK^T + online softmax; wave 4: async K staging; all: PV.
__global__ void attn_kernel(const unsigned short* __restrict__ Q,
                            const unsigned short* __restrict__ K,
                            const unsigned short* __restrict__ VT,
                            float* __restrict__ out) {
  __shared__ __align__(32) unsigned short Plds[64 * 64];   // P block (bf16)
  __shared__ __align__(32) unsigned short Klds[2][64 * 32]; // K double buffer
  __shared__ float alphaLds[64];
  __shared__ float lLds[64];
  const int lane = threadIdx.x & 31, wave = threadIdx.x >> 5;
  const int hp = lane >> 4, ln = lane & 15;
  const int qi = blockIdx.x;           // query block (64 queries)
  const int b  = blockIdx.y;           // batch
  const int qbase = qi * 64;
  const int cbase = wave * 128;        // D columns owned by this wave in PV

  f32x8 accO[32];                      // 4 query strips x 8 col tiles
#pragma unroll
  for (int i = 0; i < 32; ++i) accO[i] = zero8();

  bf16x16 aQ = zbf16();
  float m_run[8], l_run[8];
#pragma unroll
  for (int i = 0; i < 8; ++i) { m_run[i] = -3.0e38f; l_run[i] = 0.0f; }
  if (wave < 4) {
    const int qrow = qbase + wave * 16 + ln;
    const unsigned short* qr = Q + ((size_t)b * kS + qrow) * kDK;
    const int abase = hp ? 16 : 0;
    aQ = cat16(ld8(qr + abase), ld8(qr + abase + 8));
  }

  // stage K block 0
  if (wave == 4) {
    const unsigned short* kg = K + ((size_t)b * kS) * kDK;
#pragma unroll
    for (int j = 0; j < 8; ++j)
      async_ld16(kg + (j * 32 + lane) * 8, &Klds[0][(j * 32 + lane) * 8]);
  }

  for (int kb = 0; kb <= qi; ++kb) {           // causal: only blocks <= qi
    const int cur = kb & 1;
    if (wave == 4) async_wait0();              // Klds[cur] landed
    __syncthreads();                           // publish K; fences prev Plds use
    if (wave == 4 && kb < qi) {                // stage next K block
      const unsigned short* kg = K + ((size_t)b * kS + (kb + 1) * 64) * kDK;
#pragma unroll
      for (int j = 0; j < 8; ++j)
        async_ld16(kg + (j * 32 + lane) * 8, &Klds[cur ^ 1][(j * 32 + lane) * 8]);
    }
    if (wave < 4) {
      f32x8 sv[4];
#pragma unroll
      for (int nt = 0; nt < 4; ++nt) {
        bf16x16 bK = ld16(&Klds[cur][(nt * 16 + ln) * 32 + (hp ? 16 : 0)]);
        f32x8 s = wmma_bf16(aQ, bK, zero8());
#pragma unroll
        for (int i = 0; i < 8; ++i) {
          int qrow = qbase + wave * 16 + i + hp * 8;
          int kcol = kb * 64 + nt * 16 + ln;
          float v = s[i] * kInvSqrtDk;
          if (kcol > qrow) v = kNinf;          // causal mask replaces score
          sv[nt][i] = v;
        }
      }
#pragma unroll
      for (int i = 0; i < 8; ++i) {
        float rm = fmaxf(fmaxf(sv[0][i], sv[1][i]), fmaxf(sv[2][i], sv[3][i]));
#pragma unroll
        for (int m = 1; m < 16; m <<= 1) rm = fmaxf(rm, __shfl_xor(rm, m, 32));
        float mnew  = fmaxf(m_run[i], rm);
        float alpha = __expf(m_run[i] - mnew);
        m_run[i] = mnew;
        float ps = 0.0f;
#pragma unroll
        for (int nt = 0; nt < 4; ++nt) {
          float p = __expf(sv[nt][i] - mnew);
          ps += p;
          Plds[(wave * 16 + i + hp * 8) * 64 + nt * 16 + ln] = f2bfu(p);
        }
#pragma unroll
        for (int m = 1; m < 16; m <<= 1) ps += __shfl_xor(ps, m, 32);
        l_run[i] = l_run[i] * alpha + ps;
        if (ln == 0) alphaLds[wave * 16 + i + hp * 8] = alpha;
      }
    }
    __syncthreads();                           // P + alpha ready
    // PV: every wave accumulates its 64x128 output slice
#pragma unroll
    for (int q = 0; q < 4; ++q) {
      float a[8];
#pragma unroll
      for (int i = 0; i < 8; ++i) a[i] = alphaLds[q * 16 + i + hp * 8];
      const int prow = q * 16 + ln;
      const int pb = hp ? 16 : 0;
      bf16x16 aP0 = cat16(*(const bf16x8*)&Plds[prow * 64 + pb],
                          *(const bf16x8*)&Plds[prow * 64 + pb + 8]);
      bf16x16 aP1 = cat16(*(const bf16x8*)&Plds[prow * 64 + 32 + pb],
                          *(const bf16x8*)&Plds[prow * 64 + 32 + pb + 8]);
#pragma unroll
      for (int ct = 0; ct < 8; ++ct) {
        f32x8 acc = accO[q * 8 + ct];
#pragma unroll
        for (int i = 0; i < 8; ++i) acc[i] *= a[i];
        const int d = cbase + ct * 16 + ln;
        const unsigned short* vr = VT + ((size_t)b * kD + d) * kS + kb * 64;
        acc = wmma_bf16(aP0, ld16(vr +      (hp ? 16 : 0)), acc);
        acc = wmma_bf16(aP1, ld16(vr + 32 + (hp ? 16 : 0)), acc);
        accO[q * 8 + ct] = acc;
      }
    }
  }

  if (wave < 4) {
#pragma unroll
    for (int i = 0; i < 8; ++i)
      if (ln == 0) lLds[wave * 16 + i + hp * 8] = l_run[i];
  }
  __syncthreads();
#pragma unroll
  for (int q = 0; q < 4; ++q) {
    float linv[8];
#pragma unroll
    for (int i = 0; i < 8; ++i) linv[i] = 1.0f / lLds[q * 16 + i + hp * 8];
#pragma unroll
    for (int ct = 0; ct < 8; ++ct) {
#pragma unroll
      for (int i = 0; i < 8; ++i) {
        int qrow = qbase + q * 16 + i + hp * 8;
        int d = cbase + ct * 16 + ln;
        out[((size_t)b * kS + qrow) * kD + d] = accO[q * 8 + ct][i] * linv[i];
      }
    }
  }
}

extern "C" void kernel_launch(void* const* d_in, const int* in_sizes, int n_in,
                              void* d_out, int out_size, void* d_ws, size_t ws_size,
                              hipStream_t stream) {
  (void)in_sizes; (void)n_in; (void)out_size; (void)ws_size;
  const float* x  = (const float*)d_in[0];
  // d_in[1]: attn_mask — deterministic causal (triu, k=1), computed analytically
  const unsigned char* kpm = (const unsigned char*)d_in[2]; // bool [B,S,1]
  const float* Wq = (const float*)d_in[3];
  const float* Wk = (const float*)d_in[4];
  const float* Wv = (const float*)d_in[5];
  float* out = (float*)d_out;

  char* ws = (char*)d_ws;
  size_t off = 0;
  auto alloc = [&](size_t bytes) {
    void* p = ws + off;
    off = (off + bytes + 255) & ~(size_t)255;
    return p;
  };
  unsigned short* WqT = (unsigned short*)alloc((size_t)kDK * kD * 2);  // 64KB
  unsigned short* WkT = (unsigned short*)alloc((size_t)kDK * kD * 2);  // 64KB
  unsigned short* WvT = (unsigned short*)alloc((size_t)kD * kD * 2);   // 2MB
  unsigned short* Qb  = (unsigned short*)alloc((size_t)kB * kS * kDK * 2); // 1MB
  unsigned short* Kb  = (unsigned short*)alloc((size_t)kB * kS * kDK * 2); // 1MB
  unsigned short* VT  = (unsigned short*)alloc((size_t)kB * kD * kS * 2);  // 32MB

  hipLaunchKernelGGL(transpose_to_bf16, dim3(64),  dim3(256), 0, stream, Wq, WqT, kD, kDK);
  hipLaunchKernelGGL(transpose_to_bf16, dim3(64),  dim3(256), 0, stream, Wk, WkT, kD, kDK);
  hipLaunchKernelGGL(transpose_to_bf16, dim3(512), dim3(256), 0, stream, Wv, WvT, kD, kD);
  hipLaunchKernelGGL(proj_qk_kernel, dim3(128),       dim3(256), 0, stream,
                     x, WqT, WkT, kpm, Qb, Kb);
  hipLaunchKernelGGL(proj_v_kernel,  dim3(8, 128),    dim3(256), 0, stream,
                     x, WvT, VT);
  hipLaunchKernelGGL(attn_kernel,    dim3(kS / 64, kB), dim3(256), 0, stream,
                     Qb, Kb, VT, out);
}
